// BurnoutGAT_6536940224655
// MI455X (gfx1250) — compile-verified, matrix operand order checked by reading, not run on hardware
//
#include <hip/hip_runtime.h>

// ---------------------------------------------------------------------------
// BurnoutGAT on MI455X (gfx1250): 2-layer GAT + BN/ELU + linear head.
// GEMMs via v_wmma_f32_16x16x32_f16 with operands pre-packed into the exact
// WMMA fragment layout (contiguous 32B/lane -> global_load_b128 pairs) and
// tile-padded outputs so C-stores are branch-free 8-store clauses.
// Edge softmax via float atomics (the HBM-bound part of this workload).
// ---------------------------------------------------------------------------

typedef __attribute__((ext_vector_type(16))) _Float16 v16h;
typedef __attribute__((ext_vector_type(8)))  float    v8f;

// K index held by packed half `idx` (= 2*v+elem) of lane-half `half`:
//   k = (v<4 ? 2v : 16+2(v-4)) + 8*half + elem
__device__ __forceinline__ int kmap(int v, int half) {
    return (v < 4 ? 2 * v : 16 + 2 * (v - 4)) + 8 * half;
}
// Inverse: k (0..31) -> (lane-half, packed half index 0..15)
__device__ __forceinline__ void kk_to_frag(int kk, int& half, int& idx) {
    int hi   = kk >= 16;
    int r    = kk & 15;
    half     = (r >= 8) ? 1 : 0;
    int base = r - 8 * half;                 // 0..7
    int v    = (hi ? 4 : 0) + (base >> 1);
    idx      = 2 * v + (base & 1);
}

// ------------------------------- pack kernels ------------------------------
// A-pack from f32 [N, srcK] (row-major) -> frag tiles [tile][lane][16],
// K padded to 32 with zeros. (Used for x: srcK=13.)
__global__ void k_pack_a32(const float* __restrict__ src, _Float16* __restrict__ dst,
                           int N, int srcK) {
    long t = (long)blockIdx.x * blockDim.x + threadIdx.x;
    long total = (long)((N + 15) / 16) * 512;
    if (t >= total) return;
    int tile = (int)(t >> 9);
    int rem  = (int)(t & 511);
    int lane = rem >> 4, idx = rem & 15;
    int row  = lane & 15, half = lane >> 4;
    int v = idx >> 1, elem = idx & 1;
    int k = kmap(v, half) + elem;
    int n = tile * 16 + row;
    float val = (n < N && k < srcK) ? src[(size_t)n * srcK + k] : 0.f;
    dst[t] = (_Float16)val;
}

// B-pack from f32 W[Ksrc, C] -> [colTile][kStep][lane][16]; K padded to Kp.
__global__ void k_pack_b32(const float* __restrict__ W, _Float16* __restrict__ dst,
                           int Ksrc, int Kp, int C) {
    long t = (long)blockIdx.x * blockDim.x + threadIdx.x;
    long per = (long)(Kp / 32) * 512;
    long total = (long)(C / 16) * per;
    if (t >= total) return;
    int ct   = (int)(t / per);
    long r2  = t % per;
    int ks   = (int)(r2 >> 9);
    int rem  = (int)(r2 & 511);
    int lane = rem >> 4, idx = rem & 15;
    int col  = ct * 16 + (lane & 15);
    int half = lane >> 4;
    int v = idx >> 1, elem = idx & 1;
    int k = ks * 32 + kmap(v, half) + elem;
    float val = (k < Ksrc) ? W[(size_t)k * C + col] : 0.f;
    dst[t] = (_Float16)val;
}

// --------- C/D store (frag -> row-major); output is tile-padded ------------
__device__ __forceinline__ void store_c_frag(float* C, v8f c, int rowBase,
                                             int colBase, int ldc) {
    int lane = threadIdx.x & 31;
    int col  = colBase + (lane & 15);
    int half = lane >> 4;
    float* p = C + (size_t)(rowBase + 8 * half) * ldc + col;
#pragma unroll
    for (int r = 0; r < 8; ++r) p[(size_t)r * ldc] = c[r];
}

// ------------------------------- GEMM kernels ------------------------------
// h1[Np,256] = x(pk)[Np,32] @ W1(pk)[32,256]; block=128 (4 waves), 1 row tile.
__global__ void k_gemm1(const _Float16* __restrict__ apk,
                        const _Float16* __restrict__ bpk,
                        float* __restrict__ h1) {
    int wave = threadIdx.x >> 5;
    int lane = threadIdx.x & 31;
    int tile = blockIdx.x;
    v16h a = ((const v16h*)(apk + (size_t)tile * 512))[lane];
#pragma unroll
    for (int i = 0; i < 4; ++i) {                    // 16 col tiles / 4 waves
        int t = wave + 4 * i;
        v16h b = ((const v16h*)(bpk + (size_t)t * 512))[lane];
        v8f  c = {};
        c = __builtin_amdgcn_wmma_f32_16x16x32_f16(false, a, false, b,
                                                   (short)0, c, false, false);
        store_c_frag(h1, c, tile * 16, t * 16, 256);
    }
}

// h2[Np,64] = hbn(pk)[Np,256] @ W2(pk)[256,64]; block=128, wave = col tile.
__global__ void k_gemm2(const _Float16* __restrict__ apk,
                        const _Float16* __restrict__ bpk,
                        float* __restrict__ out) {
    int wave = threadIdx.x >> 5;
    int lane = threadIdx.x & 31;
    int tile = blockIdx.x;
    v8f c = {};
#pragma unroll
    for (int kk = 0; kk < 8; ++kk) {                 // K = 8 * 32 = 256
        v16h a = ((const v16h*)(apk + ((size_t)tile * 8 + kk) * 512))[lane];
        v16h b = ((const v16h*)(bpk + ((size_t)wave * 8 + kk) * 512))[lane];
        c = __builtin_amdgcn_wmma_f32_16x16x32_f16(false, a, false, b,
                                                   (short)0, c, false, false);
    }
    store_c_frag(out, c, tile * 16, wave * 16, 64);
}

// ------------------------------ small kernels ------------------------------
__global__ void k_fill(float* p, float v, long n) {
    long i = (long)blockIdx.x * blockDim.x + threadIdx.x;
    if (i < n) p[i] = v;
}

// a_src[n,h], a_dst[n,h] = sum_c feat[n,h,c] * att_{src,dst}[h,c]
template <int H, int C>
__global__ void k_att(const float* __restrict__ feat,
                      const float* __restrict__ att_s,
                      const float* __restrict__ att_d,
                      float* __restrict__ as_, float* __restrict__ ad_, int N) {
    int id = blockIdx.x * blockDim.x + threadIdx.x;
    if (id >= N * H) return;
    int n = id / H, h = id % H;
    const float* f = feat + ((size_t)n * H + h) * C;
    float s = 0.f, d = 0.f;
#pragma unroll 8
    for (int c = 0; c < C; ++c) {
        s += f[c] * att_s[h * C + c];
        d += f[c] * att_d[h * C + c];
    }
    as_[id] = s;
    ad_[id] = d;
}

__device__ __forceinline__ void atomicMaxFloat(float* addr, float v) {
    if (v >= 0.f) atomicMax((int*)addr, __float_as_int(v));
    else          atomicMin((unsigned int*)addr, __float_as_uint(v));
}

__device__ __forceinline__ void edge_sd(const int* ei, int E, int e, int& s, int& d) {
    if (e < E) { s = ei[e]; d = ei[E + e]; }
    else       { s = d = e - E; }                     // self loop
}

template <int H>
__global__ void k_edge_max(const int* __restrict__ ei, int E, int ET,
                           const float* __restrict__ asrc,
                           const float* __restrict__ adst,
                           float* __restrict__ amax) {
    int e = blockIdx.x * blockDim.x + threadIdx.x;
    if (e >= ET) return;
    int s, d; edge_sd(ei, E, e, s, d);
#pragma unroll
    for (int h = 0; h < H; ++h) {
        float a = asrc[s * H + h] + adst[d * H + h];
        a = a > 0.f ? a : 0.2f * a;                   // leaky_relu(0.2)
        atomicMaxFloat(&amax[d * H + h], a);
    }
}

template <int H>
__global__ void k_edge_exp(const int* __restrict__ ei, int E, int ET,
                           const float* __restrict__ asrc,
                           const float* __restrict__ adst,
                           const float* __restrict__ amax,
                           float* __restrict__ walpha,
                           float* __restrict__ denom) {
    int e = blockIdx.x * blockDim.x + threadIdx.x;
    if (e >= ET) return;
    int s, d; edge_sd(ei, E, e, s, d);
#pragma unroll
    for (int h = 0; h < H; ++h) {
        float a = asrc[s * H + h] + adst[d * H + h];
        a = a > 0.f ? a : 0.2f * a;
        float w = __expf(a - amax[d * H + h]);
        walpha[(size_t)e * H + h] = w;
        atomicAdd(&denom[d * H + h], w);
    }
}

// out[dst,h,c] += feat[src,h,c] * walpha[e,h]/denom[dst,h]; one thread/(e,c)
template <int H, int C>
__global__ void k_edge_aggr(const int* __restrict__ ei, int E, int ET,
                            const float* __restrict__ feat,
                            const float* __restrict__ walpha,
                            const float* __restrict__ denom,
                            float* __restrict__ out) {
    long t = (long)blockIdx.x * blockDim.x + threadIdx.x;
    if (t >= (long)ET * C) return;
    int e = (int)(t / C), c = (int)(t % C);           // C is power of two
    int s, d; edge_sd(ei, E, e, s, d);
    __builtin_prefetch(&feat[(size_t)s * H * C + c], 0, 0);   // global_prefetch
#pragma unroll
    for (int h = 0; h < H; ++h) {
        float w = walpha[(size_t)e * H + h] / denom[d * H + h];
        atomicAdd(&out[((size_t)d * H + h) * C + c],
                  feat[((size_t)s * H + h) * C + c] * w);
    }
}

// Per-channel sum / sum-of-squares for BatchNorm (256 channels).
__global__ void k_bn_stats(const float* __restrict__ out1,
                           float* __restrict__ bn_sum,
                           float* __restrict__ bn_sumsq, int N) {
    int c = threadIdx.x;                               // 0..255
    int chunk = (N + gridDim.x - 1) / gridDim.x;
    int n0 = blockIdx.x * chunk;
    int n1 = n0 + chunk; if (n1 > N) n1 = N;
    float s = 0.f, s2 = 0.f;
    for (int n = n0; n < n1; ++n) {
        float v = out1[(size_t)n * 256 + c];
        s += v; s2 += v * v;
    }
    atomicAdd(&bn_sum[c], s);
    atomicAdd(&bn_sumsq[c], s2);
}

// BN (batch stats) + ELU; writes fp16 DIRECTLY in WMMA A-fragment layout
// [tile][kstep][lane][16] for GEMM2. (b1 is absorbed by BatchNorm.)
__global__ void k_bn_elu_pack(const float* __restrict__ out1,
                              const float* __restrict__ bn_sum,
                              const float* __restrict__ bn_sumsq,
                              const float* __restrict__ gamma,
                              const float* __restrict__ beta,
                              _Float16* __restrict__ apk, int N) {
    long t = (long)blockIdx.x * blockDim.x + threadIdx.x;
    if (t >= (long)N * 256) return;
    int n = (int)(t >> 8), c = (int)(t & 255);
    float inv = 1.f / (float)N;
    float mu  = bn_sum[c] * inv;
    float var = bn_sumsq[c] * inv - mu * mu;
    float y = (out1[t] - mu) * rsqrtf(var + 1e-5f) * gamma[c] + beta[c];
    y = y > 0.f ? y : __expf(y) - 1.f;                 // ELU
    // packed destination: tile = n/16, row = n%16, kstep = c/32, kk = c%32
    int tile = n >> 4, row = n & 15, ks = c >> 5, kk = c & 31;
    int half, idx;
    kk_to_frag(kk, half, idx);
    int lane = row + 16 * half;
    apk[(((size_t)tile * 8 + ks) * 32 + lane) * 16 + idx] = (_Float16)y;
}

// head-mean + b2 + ELU + dot(Wc) + bc ; one wave32 per node.
__global__ void k_final(const float* __restrict__ out2,
                        const float* __restrict__ b2,
                        const float* __restrict__ Wc,
                        const float* __restrict__ bc,
                        float* __restrict__ out, int N) {
    int gid  = blockIdx.x * blockDim.x + threadIdx.x;
    int wid  = gid >> 5;
    int lane = threadIdx.x & 31;
    if (wid >= N) return;
    float v = 0.5f * (out2[(size_t)wid * 64 + lane] +
                      out2[(size_t)wid * 64 + 32 + lane]) + b2[lane];
    v = v > 0.f ? v : __expf(v) - 1.f;                 // ELU
    float p = v * Wc[lane];
    for (int off = 16; off; off >>= 1) p += __shfl_down(p, off, 32);
    if (lane == 0) out[wid] = p + bc[0];
}

// --------------------------------- launch ----------------------------------
extern "C" void kernel_launch(void* const* d_in, const int* in_sizes, int n_in,
                              void* d_out, int out_size, void* d_ws, size_t ws_size,
                              hipStream_t stream) {
    const float* x        = (const float*)d_in[0];
    const int*   ei       = (const int*)  d_in[1];
    const float* W1       = (const float*)d_in[2];
    const float* att_src1 = (const float*)d_in[3];
    const float* att_dst1 = (const float*)d_in[4];
    // d_in[5] = b1 (absorbed by BatchNorm: (h+b1-mean(h+b1)) == (h-mean(h)))
    const float* gamma1   = (const float*)d_in[6];
    const float* beta1    = (const float*)d_in[7];
    const float* W2       = (const float*)d_in[8];
    const float* att_src2 = (const float*)d_in[9];
    const float* att_dst2 = (const float*)d_in[10];
    const float* b2       = (const float*)d_in[11];
    const float* Wc       = (const float*)d_in[12];
    const float* bc       = (const float*)d_in[13];
    float* out = (float*)d_out;

    const int N  = in_sizes[0] / 13;      // 50000
    const int E  = in_sizes[1] / 2;       // 800000
    const int ET = E + N;                 // + self loops
    const int NT = (N + 15) / 16;         // row tiles
    const int Np = NT * 16;               // tile-padded row count

    // ---- workspace carve-out ----
    size_t off = 0;
    auto carve = [&](size_t bytes) -> void* {
        void* p = (char*)d_ws + off;
        off += (bytes + 255) & ~(size_t)255;
        return p;
    };
    float*    h1      = (float*)   carve((size_t)Np * 256 * 4);       // padded
    float*    out1    = (float*)   carve((size_t)N * 256 * 4);
    float*    a_src1  = (float*)   carve((size_t)N * 4 * 4);
    float*    a_dst1  = (float*)   carve((size_t)N * 4 * 4);
    float*    amax1   = (float*)   carve((size_t)N * 4 * 4);
    float*    denom1  = (float*)   carve((size_t)N * 4 * 4);
    float*    walpha1 = (float*)   carve((size_t)ET * 4 * 4);
    float*    bn_sum  = (float*)   carve(256 * 4);
    float*    bn_sq   = (float*)   carve(256 * 4);
    _Float16* apk1    = (_Float16*)carve((size_t)NT * 512 * 2);       // x packed
    _Float16* bpk1    = (_Float16*)carve((size_t)16 * 512 * 2);       // W1 packed
    _Float16* apk2    = (_Float16*)carve((size_t)NT * 8 * 512 * 2);   // hbn packed
    _Float16* bpk2    = (_Float16*)carve((size_t)4 * 8 * 512 * 2);    // W2 packed
    float*    walpha2 = (float*)   carve((size_t)ET * 2 * 4);
    // layer-2 buffers reuse h1's region (free after layer-1 aggregation)
    float* h2     = h1;                        // Np*64 (padded GEMM output)
    float* out2   = h1 + (size_t)Np * 64;      // N*64
    float* a_src2 = h1 + (size_t)Np * 128;     // N*2
    float* a_dst2 = h1 + (size_t)Np * 130;     // N*2
    float* amax2  = h1 + (size_t)Np * 132;     // N*2
    float* denom2 = h1 + (size_t)Np * 134;     // N*2

    const int B = 256;
    auto blocks = [&](long n) { return (int)((n + B - 1) / B); };
    const float NEG = -3.402823466e38f;

    // ---- pack WMMA operands ----
    k_pack_a32<<<blocks((long)NT * 512), B, 0, stream>>>(x, apk1, N, 13);
    k_pack_b32<<<blocks(16L * 512),      B, 0, stream>>>(W1, bpk1, 13, 32, 256);
    k_pack_b32<<<blocks(4L * 8 * 512),   B, 0, stream>>>(W2, bpk2, 256, 256, 64);

    // ---- layer-1 init ----
    k_fill<<<blocks((long)N * 256), B, 0, stream>>>(out1, 0.f, (long)N * 256);
    k_fill<<<blocks((long)N * 4),   B, 0, stream>>>(amax1, NEG, (long)N * 4);
    k_fill<<<blocks((long)N * 4),   B, 0, stream>>>(denom1, 0.f, (long)N * 4);
    k_fill<<<blocks(256),           B, 0, stream>>>(bn_sum, 0.f, 256);
    k_fill<<<blocks(256),           B, 0, stream>>>(bn_sq, 0.f, 256);

    // ---- layer 1: GEMM (WMMA) + attention + edge softmax + aggregate ----
    k_gemm1<<<NT, 128, 0, stream>>>(apk1, bpk1, h1);
    k_att<4, 64><<<blocks((long)N * 4), B, 0, stream>>>(h1, att_src1, att_dst1,
                                                        a_src1, a_dst1, N);
    k_edge_max<4><<<blocks(ET), B, 0, stream>>>(ei, E, ET, a_src1, a_dst1, amax1);
    k_edge_exp<4><<<blocks(ET), B, 0, stream>>>(ei, E, ET, a_src1, a_dst1, amax1,
                                                walpha1, denom1);
    k_edge_aggr<4, 64><<<blocks((long)ET * 64), B, 0, stream>>>(ei, E, ET, h1,
                                                                walpha1, denom1, out1);

    // ---- BatchNorm stats, then re-init layer-2 scratch (h1 now free) ----
    k_bn_stats<<<128, 256, 0, stream>>>(out1, bn_sum, bn_sq, N);
    k_fill<<<blocks((long)N * 64), B, 0, stream>>>(out2, 0.f, (long)N * 64);
    k_fill<<<blocks((long)N * 2),  B, 0, stream>>>(amax2, NEG, (long)N * 2);
    k_fill<<<blocks((long)N * 2),  B, 0, stream>>>(denom2, 0.f, (long)N * 2);
    k_bn_elu_pack<<<blocks((long)N * 256), B, 0, stream>>>(out1, bn_sum, bn_sq,
                                                           gamma1, beta1, apk2, N);

    // ---- layer 2: GEMM (WMMA, K=256) + attention + edge softmax + agg ----
    k_gemm2<<<NT, 128, 0, stream>>>(apk2, bpk2, h2);
    k_att<2, 32><<<blocks((long)N * 2), B, 0, stream>>>(h2, att_src2, att_dst2,
                                                        a_src2, a_dst2, N);
    k_edge_max<2><<<blocks(ET), B, 0, stream>>>(ei, E, ET, a_src2, a_dst2, amax2);
    k_edge_exp<2><<<blocks(ET), B, 0, stream>>>(ei, E, ET, a_src2, a_dst2, amax2,
                                                walpha2, denom2);
    k_edge_aggr<2, 32><<<blocks((long)ET * 32), B, 0, stream>>>(ei, E, ET, h2,
                                                                walpha2, denom2, out2);

    // ---- head mean + b2 + ELU + Wc projection ----
    k_final<<<blocks((long)N * 32), B, 0, stream>>>(out2, b2, Wc, bc, out, N);
}
